// SpikingLocusCoeruleus_38869454029215
// MI455X (gfx1250) — compile-verified, matrix operand order checked by reading, not run on hardware
//
#include <hip/hip_runtime.h>

// ---------------------------------------------------------------------------
// SpikingLocusCoeruleus on MI455X (gfx1250, wave32)
// Memory-bound streaming kernel: 20 Euler steps over 4M Izhikevich neurons.
// float4 (b128) coalesced noise loads, state in registers, WMMA wave-reduce.
// ---------------------------------------------------------------------------

typedef float    v2f  __attribute__((ext_vector_type(2)));
typedef float    v8f  __attribute__((ext_vector_type(8)));
typedef _Float16 v16h __attribute__((ext_vector_type(16)));

// --- Izhikevich 'RS' update, matches reference order exactly -----------------
__device__ __forceinline__ void izh_step(float& v, float& u, float& r,
                                         float Ic, float nz) {
    const float I = Ic + 0.5f * nz;                        // I_drive + i_tonic + 0.5*noise
    v = v + (0.04f * v * v + 5.0f * v + 140.0f - u + I);   // dt=1 Euler
    u = u + 0.02f * (0.2f * v - u);                        // uses updated v
    const bool sp = (v >= 30.0f);
    if (sp) { v = -65.0f; u = u + 8.0f; }
    r = 0.9f * r + 0.1f * (sp ? 1.0f : 0.0f);
}

// --- wave32 sum via WMMA: D = A(16x4 partials) x B(4x16 ones) ----------------
// A layout (f32 16x4): lanes 0-15 hold {K0,K1}, lanes 16-31 hold {K2,K3}.
// With a = {partial, 0}: rowsum_m = s[m] + s[m+16], replicated over all N.
// Lane<16 D regs hold rows 0..7, lane>=16 hold rows 8..15 -> fold + shfl_xor16.
__device__ __forceinline__ float wave_reduce_wmma(float partial) {
#if __has_builtin(__builtin_amdgcn_wmma_f32_16x16x4_f32)
    v2f a; a[0] = partial; a[1] = 0.0f;
    v2f b; b[0] = 1.0f;    b[1] = 1.0f;
    v8f c = {};
    v8f d = __builtin_amdgcn_wmma_f32_16x16x4_f32(
        /*neg_a=*/false, a, /*neg_b=*/false, b,
        /*c_mod=*/(short)0, c, /*reuse_a=*/false, /*reuse_b=*/false);
    float local = ((d[0] + d[1]) + (d[2] + d[3])) + ((d[4] + d[5]) + (d[6] + d[7]));
    return local + __shfl_xor(local, 16, 32);
#else
    // fallback: confirmed f16 WMMA; two-term f16 split keeps ~f32 precision
    v16h a = {};
    _Float16 hi = (_Float16)partial;
    a[0] = hi;
    a[1] = (_Float16)(partial - (float)hi);
    v16h b;
    #pragma unroll
    for (int i = 0; i < 16; ++i) b[i] = (_Float16)1.0f;
    v8f c = {};
    v8f d = __builtin_amdgcn_wmma_f32_16x16x32_f16(
        false, a, false, b, (short)0, c, false, false);
    float local = ((d[0] + d[1]) + (d[2] + d[3])) + ((d[4] + d[5]) + (d[6] + d[7]));
    return local + __shfl_xor(local, 16, 32);
#endif
}

// --- zero a single float (for accumulator / tail slot) -----------------------
__global__ void lc_zero_one(float* __restrict__ p) { *p = 0.0f; }

// --- main streaming kernel: 256 threads x 4 neurons each ---------------------
// No early exit: EXEC must stay all-ones for WMMA; grid sized exactly.
__global__ __launch_bounds__(256)
void lc_main(const float* __restrict__ amyg, const float* __restrict__ insula,
             const float* __restrict__ circ, const float* __restrict__ cms,
             const float* __restrict__ v0,   const float* __restrict__ u0,
             const float* __restrict__ r0,   const float* __restrict__ noise,
             float* __restrict__ out_sums, int N, int T, int det)
{
    const float tonic = 2.0f + circ[0] * 3.0f + insula[0] * 5.0f + cms[0] * 3.0f;
    const float Ic    = tonic + ((amyg[0] > 0.3f) ? 15.0f : 0.0f) + 0.5f;

    const int  tid  = blockIdx.x * 256 + threadIdx.x;
    const int  base = tid * 4;

    float4 v = *(const float4*)(v0 + base);
    float4 u = *(const float4*)(u0 + base);
    float4 r = *(const float4*)(r0 + base);

    for (int t = 0; t < T; ++t) {
        const float4 nz = *(const float4*)(noise + (size_t)t * (size_t)N + base);
        if (t + 1 < T)  // pull next 16MB slab toward L2 while this one computes
            __builtin_prefetch(noise + (size_t)(t + 1) * (size_t)N + base, 0, 1);
        izh_step(v.x, u.x, r.x, Ic, nz.x);
        izh_step(v.y, u.y, r.y, Ic, nz.y);
        izh_step(v.z, u.z, r.z, Ic, nz.z);
        izh_step(v.w, u.w, r.w, Ic, nz.w);
    }

    const float partial = (r.x + r.y) + (r.z + r.w);
    const float wsum    = wave_reduce_wmma(partial);   // all lanes hold wave sum

    __shared__ float smem[8];
    const int lane = threadIdx.x & 31;
    const int wid  = threadIdx.x >> 5;
    if (lane == 0) smem[wid] = wsum;
    __syncthreads();
    if (threadIdx.x == 0) {
        float bs = 0.0f;
        #pragma unroll
        for (int i = 0; i < 8; ++i) bs += smem[i];
        if (det) out_sums[blockIdx.x] = bs;            // deterministic path
        else     atomicAdd(out_sums, bs);              // tiny-ws fallback
    }
}

// --- scalar tail (only if N % 1024 != 0; unused at N = 4M) -------------------
__global__ void lc_tail(const float* __restrict__ amyg, const float* __restrict__ insula,
                        const float* __restrict__ circ, const float* __restrict__ cms,
                        const float* __restrict__ v0,   const float* __restrict__ u0,
                        const float* __restrict__ r0,   const float* __restrict__ noise,
                        float* __restrict__ acc, int N, int T, int start)
{
    const int i = start + blockIdx.x * 256 + threadIdx.x;
    if (i >= N) return;
    const float tonic = 2.0f + circ[0] * 3.0f + insula[0] * 5.0f + cms[0] * 3.0f;
    const float Ic    = tonic + ((amyg[0] > 0.3f) ? 15.0f : 0.0f) + 0.5f;
    float v = v0[i], u = u0[i], r = r0[i];
    for (int t = 0; t < T; ++t)
        izh_step(v, u, r, Ic, noise[(size_t)t * (size_t)N + i]);
    atomicAdd(acc, r);
}

// --- deterministic final reduce + noradrenaline readout ----------------------
__global__ __launch_bounds__(256)
void lc_finalize_det(const float* __restrict__ sums, int nsum,
                     const float* __restrict__ amyg,
                     float* __restrict__ out, float invN)
{
    __shared__ float s[256];
    float acc = 0.0f;
    for (int i = threadIdx.x; i < nsum; i += 256) acc += sums[i];  // fixed order
    s[threadIdx.x] = acc;
    __syncthreads();
    for (int off = 128; off > 0; off >>= 1) {
        if (threadIdx.x < off) s[threadIdx.x] += s[threadIdx.x + off];
        __syncthreads();
    }
    if (threadIdx.x == 0) {
        const float lc_mean = s[0] * invN;
        float raw = fminf(fmaxf(lc_mean * 6.0f, 0.0f), 1.0f);
        if (amyg[0] > 0.3f) raw = fminf(raw, 0.8f);
        float na = 0.85f * 0.3f + 0.15f * raw;
        na = fminf(fmaxf(na, 0.05f), 0.95f);
        out[0] = na;
        out[1] = lc_mean;
        out[2] = fminf(na * 2.0f, 1.0f);
        out[3] = fminf(na * 1.5f, 1.0f);
    }
}

__global__ void lc_finalize_scalar(const float* __restrict__ acc,
                                   const float* __restrict__ amyg,
                                   float* __restrict__ out, float invN)
{
    const float lc_mean = acc[0] * invN;
    float raw = fminf(fmaxf(lc_mean * 6.0f, 0.0f), 1.0f);
    if (amyg[0] > 0.3f) raw = fminf(raw, 0.8f);
    float na = 0.85f * 0.3f + 0.15f * raw;
    na = fminf(fmaxf(na, 0.05f), 0.95f);
    out[0] = na;
    out[1] = lc_mean;
    out[2] = fminf(na * 2.0f, 1.0f);
    out[3] = fminf(na * 1.5f, 1.0f);
}

extern "C" void kernel_launch(void* const* d_in, const int* in_sizes, int n_in,
                              void* d_out, int out_size, void* d_ws, size_t ws_size,
                              hipStream_t stream) {
    const float* amyg   = (const float*)d_in[0];
    const float* insula = (const float*)d_in[1];
    const float* circ   = (const float*)d_in[2];
    const float* cms    = (const float*)d_in[3];
    const float* v0     = (const float*)d_in[4];
    const float* u0     = (const float*)d_in[5];
    const float* r0     = (const float*)d_in[6];
    const float* noise  = (const float*)d_in[7];
    float*       out    = (float*)d_out;

    const int N = in_sizes[4];                 // 4194304
    const int T = (N > 0) ? in_sizes[7] / N : 0; // 20
    const int nblocks    = N / 1024;           // 256 thr * 4 neurons
    const int tail_start = nblocks * 1024;
    const float invN = (N > 0) ? (1.0f / (float)N) : 0.0f;

    const bool det = ws_size >= (size_t)(nblocks + 1) * sizeof(float);

    if (det) {
        float* sums = (float*)d_ws;
        lc_zero_one<<<1, 1, 0, stream>>>(sums + nblocks);   // tail slot
        if (nblocks > 0)
            lc_main<<<nblocks, 256, 0, stream>>>(amyg, insula, circ, cms,
                                                 v0, u0, r0, noise,
                                                 sums, N, T, 1);
        if (tail_start < N) {
            const int tb = (N - tail_start + 255) / 256;
            lc_tail<<<tb, 256, 0, stream>>>(amyg, insula, circ, cms,
                                            v0, u0, r0, noise,
                                            sums + nblocks, N, T, tail_start);
        }
        lc_finalize_det<<<1, 256, 0, stream>>>(sums, nblocks + 1, amyg, out, invN);
    } else {
        float* acc = (ws_size >= sizeof(float)) ? (float*)d_ws : (out + 1);
        lc_zero_one<<<1, 1, 0, stream>>>(acc);
        if (nblocks > 0)
            lc_main<<<nblocks, 256, 0, stream>>>(amyg, insula, circ, cms,
                                                 v0, u0, r0, noise,
                                                 acc, N, T, 0);
        if (tail_start < N) {
            const int tb = (N - tail_start + 255) / 256;
            lc_tail<<<tb, 256, 0, stream>>>(amyg, insula, circ, cms,
                                            v0, u0, r0, noise,
                                            acc, N, T, tail_start);
        }
        lc_finalize_scalar<<<1, 1, 0, stream>>>(acc, amyg, out, invN);
    }
}
